// VectorQuantizer_76295799046539
// MI455X (gfx1250) — matrix operand order repaired and checked by reading, not compiled, not run on hardware
//
#include <hip/hip_runtime.h>
#include <hip/hip_bf16.h>

typedef __attribute__((ext_vector_type(16))) _Float16 v16h;
typedef __attribute__((ext_vector_type(8)))  _Float16 v8h;
typedef __attribute__((ext_vector_type(8)))  float    v8f;

#define N_TOK   32768   // B*h*w = 32*32*32
#define C_DIM   64
#define V_CODES 8192
#define HW      1024    // h*w

// Workspace layout (bytes)
#define WS_Q_OFF    0u          // q_f16:  32768*64*2 = 4 MiB
#define WS_WN_OFF   4194304u    // wn_f16:  8192*64*2 = 1 MiB
#define WS_IDX_OFF  5242880u    // idx:    32768*4
#define WS_CNT_OFF  5373952u    // counts:  8192*4
#define WS_LOSS_OFF 5406720u    // loss accumulator: 4 B

#define CHUNK_CODES 128
#define CHUNK_BYTES (CHUNK_CODES * C_DIM * 2)   // 16 KiB

// ---------------------------------------------------------------------------
// Zero the per-call accumulators (harness does NOT re-poison between replays).
__global__ void vq_zero(unsigned* counts, float* loss_sum) {
    int i = blockIdx.x * blockDim.x + threadIdx.x;
    if (i < V_CODES) counts[i] = 0u;
    if (i == 0) loss_sum[0] = 0.0f;
}

// ---------------------------------------------------------------------------
// L2-normalize tokens: f is [B,C,h,w]; token n=(b,hw) gathers C=64 strided
// floats, normalizes, writes row-major f16 [N,64]. One wave per token.
__global__ void vq_norm_tokens(const float* __restrict__ f,
                               _Float16* __restrict__ q) {
    const int wave = threadIdx.x >> 5;
    const int lane = threadIdx.x & 31;
    const int n    = blockIdx.x * 8 + wave;
    const int b    = n >> 10;
    const int rem  = n & 1023;
    const float* fp = f + (size_t)b * C_DIM * HW + rem;
    float v0 = fp[(size_t)lane * HW];
    float v1 = fp[(size_t)(lane + 32) * HW];
    float ss = v0 * v0 + v1 * v1;
    #pragma unroll
    for (int off = 16; off; off >>= 1) ss += __shfl_xor(ss, off, 32);
    float scale = 1.0f / fmaxf(sqrtf(ss), 1e-12f);
    q[(size_t)n * C_DIM + lane]      = (_Float16)(v0 * scale);
    q[(size_t)n * C_DIM + lane + 32] = (_Float16)(v1 * scale);
}

// L2-normalize codebook rows [V,64] -> f16 [V,64]. One wave per code.
__global__ void vq_norm_codes(const float* __restrict__ emb,
                              _Float16* __restrict__ wn) {
    const int wave = threadIdx.x >> 5;
    const int lane = threadIdx.x & 31;
    const int v    = blockIdx.x * 8 + wave;
    const float* ep = emb + (size_t)v * C_DIM;
    float v0 = ep[lane];
    float v1 = ep[lane + 32];
    float ss = v0 * v0 + v1 * v1;
    #pragma unroll
    for (int off = 16; off; off >>= 1) ss += __shfl_xor(ss, off, 32);
    float scale = 1.0f / fmaxf(sqrtf(ss), 1e-12f);
    wn[(size_t)v * C_DIM + lane]      = (_Float16)(v0 * scale);
    wn[(size_t)v * C_DIM + lane + 32] = (_Float16)(v1 * scale);
}

// ---------------------------------------------------------------------------
// Async global->LDS staging of one 16 KiB codebook chunk (256 threads x 4
// b128 copies). Fire-and-forget; tracked by ASYNCcnt.
__device__ __forceinline__ void stage_chunk_async(unsigned lds_off,
                                                  const _Float16* src,
                                                  int tid) {
    const char* g = (const char*)src + tid * 16;
    unsigned    l = lds_off + tid * 16;
    #pragma unroll
    for (int i = 0; i < 4; ++i) {
        asm volatile("global_load_async_to_lds_b128 %0, %1, off"
                     :: "v"(l), "v"(g) : "memory");
        g += 256 * 16;
        l += 256 * 16;
    }
}

__device__ __forceinline__ void wait_async_done() {
    asm volatile("s_wait_asynccnt 0x0" ::: "memory");
}

// ---------------------------------------------------------------------------
// Fused GEMM + argmax. 256 threads = 8 waves; each wave owns 16 token rows
// (A fragments resident in VGPRs for the whole V loop). The workgroup
// double-buffers 128-code (16 KiB) chunks of the f16 codebook in LDS via
// async direct-to-LDS loads; each wave runs 16x16x32 WMMAs (two per code
// tile, K=64) and tracks a running per-lane (max, argmax) over the f32 C
// fragment. The sim matrix is never materialized.
__global__ void __launch_bounds__(256, 2)
vq_argmax_wmma(const _Float16* __restrict__ q,
               const _Float16* __restrict__ wn,
               int* __restrict__ idxN) {
    __shared__ _Float16 lds_w[2][CHUNK_CODES * C_DIM];   // 2 x 16 KiB

    const int tid    = threadIdx.x;
    const int lane   = tid & 31;
    const int wave   = tid >> 5;
    const int row0   = blockIdx.x * 128 + wave * 16;
    const int m      = lane & 15;
    const int halfid = lane >> 4;             // 0: lanes 0-15, 1: lanes 16-31

    const unsigned lds_base = (unsigned)(uintptr_t)(&lds_w[0][0]);

    // Kick off async stage of chunk 0 into buffer 0 (overlaps A-frag loads).
    stage_chunk_async(lds_base, wn, tid);

    // A fragments for K=0..31 (a0) and K=32..63 (a1), contiguous K order (matches B):
    // lane half 0 holds K 0..15; half 1 holds K 16..31.
    const _Float16* qp = q + (size_t)(row0 + m) * C_DIM + halfid * 16;
    v16h a0, a1;
    {
        v8h l0 = *(const v8h*)(qp);        // K  0.. 7 (+16*halfid)
        v8h h0 = *(const v8h*)(qp + 8);    // K  8..15 (+16*halfid)
        v8h l1 = *(const v8h*)(qp + 32);   // K 32..39 (+16*halfid)
        v8h h1 = *(const v8h*)(qp + 40);   // K 40..47 (+16*halfid)
        #pragma unroll
        for (int i = 0; i < 8; ++i) {
            a0[i] = l0[i]; a0[i + 8] = h0[i];
            a1[i] = l1[i]; a1[i + 8] = h1[i];
        }
    }

    float maxv[8];
    int   maxi[8];
    #pragma unroll
    for (int r = 0; r < 8; ++r) { maxv[r] = -3.4e38f; maxi[r] = 0; }

    const int koff = halfid * 16;   // B-fragment K offset within a k-tile

    wait_async_done();
    __syncthreads();

    int cur = 0;
    for (int chunk = 0; chunk < V_CODES; chunk += CHUNK_CODES) {
        // Prefetch next chunk into the other buffer while we compute.
        if (chunk + CHUNK_CODES < V_CODES) {
            stage_chunk_async(lds_base + (cur ^ 1) * CHUNK_BYTES,
                              wn + (size_t)(chunk + CHUNK_CODES) * C_DIM, tid);
        }

        const _Float16* buf = &lds_w[cur][0];
        #pragma unroll
        for (int cb = 0; cb < CHUNK_CODES; cb += 16) {
            // B fragment: lane holds column n = lane&15; half selects K 0..15
            // vs 16..31 within each k-tile -> 32 contiguous bytes per lane.
            const _Float16* wp = buf + (cb + m) * C_DIM;
            v16h b0 = *(const v16h*)(wp + koff);        // K  0..31 tile
            v16h b1 = *(const v16h*)(wp + 32 + koff);   // K 32..63 tile
            v8f c = {0.f, 0.f, 0.f, 0.f, 0.f, 0.f, 0.f, 0.f};
            c = __builtin_amdgcn_wmma_f32_16x16x32_f16(
                    false, a0, false, b0, (short)0, c, false, false);
            c = __builtin_amdgcn_wmma_f32_16x16x32_f16(
                    false, a1, false, b1, (short)0, c, false, false);
            const int code = chunk + cb + m;
            #pragma unroll
            for (int r = 0; r < 8; ++r) {
                if (c[r] > maxv[r]) { maxv[r] = c[r]; maxi[r] = code; }
            }
        }

        // Own async copies done + all waves done reading before buffer swap.
        wait_async_done();
        __syncthreads();
        cur ^= 1;
    }

    // Cross-lane argmax within each 16-lane half (C layout: lane = N column,
    // VGPR r = row M + 8*halfid). Tie-break to the lowest index (jnp.argmax).
    #pragma unroll
    for (int r = 0; r < 8; ++r) {
        float v = maxv[r];
        int   i = maxi[r];
        #pragma unroll
        for (int off = 8; off; off >>= 1) {
            float ov = __shfl_xor(v, off, 16);
            int   oi = __shfl_xor(i, off, 16);
            if (ov > v || (ov == v && oi < i)) { v = ov; i = oi; }
        }
        maxv[r] = v; maxi[r] = i;
    }
    if (m == 0) {
        const int base = row0 + halfid * 8;
        #pragma unroll
        for (int r = 0; r < 8; ++r) idxN[base + r] = maxi[r];
    }
}

// ---------------------------------------------------------------------------
// Gather codes -> output (STE output == fhat numerically, BChw layout),
// accumulate sum((fhat-f)^2), and bincount code usage.
__global__ void vq_gather_loss(const float* __restrict__ f,
                               const float* __restrict__ emb,
                               const int* __restrict__ idxN,
                               float* __restrict__ out,
                               unsigned* __restrict__ counts,
                               float* __restrict__ loss_sum) {
    const size_t gid = (size_t)blockIdx.x * blockDim.x + threadIdx.x; // over B*C*HW
    const int rem = (int)(gid & 1023);
    const int c   = (int)((gid >> 10) & 63);
    const int b   = (int)(gid >> 16);
    const int n   = b * 1024 + rem;
    const int idx = idxN[n];
    const float fh = emb[(size_t)idx * C_DIM + c];
    const float fv = f[gid];
    out[gid] = fh;
    const float d = fh - fv;

    __shared__ float red[256];
    red[threadIdx.x] = d * d;
    __syncthreads();
    #pragma unroll
    for (int s = 128; s; s >>= 1) {
        if ((int)threadIdx.x < s) red[threadIdx.x] += red[threadIdx.x + s];
        __syncthreads();
    }
    if (threadIdx.x == 0) atomicAdd(loss_sum, red[0]);
    if (c == 0) atomicAdd(&counts[idx], 1u);
}

// ---------------------------------------------------------------------------
// Scalars: vq_loss = 1.25 * mean(d^2); entropy = 0; usage = 100*used/V
// (prob > 0.01/V  <=>  count >= 1 for these sizes).
__global__ void vq_finalize(const unsigned* __restrict__ counts,
                            const float* __restrict__ loss_sum,
                            float* __restrict__ out3) {
    __shared__ int red[256];
    int used = 0;
    for (int v = threadIdx.x; v < V_CODES; v += 256) used += (counts[v] > 0u);
    red[threadIdx.x] = used;
    __syncthreads();
    #pragma unroll
    for (int s = 128; s; s >>= 1) {
        if ((int)threadIdx.x < s) red[threadIdx.x] += red[threadIdx.x + s];
        __syncthreads();
    }
    if (threadIdx.x == 0) {
        out3[0] = 1.25f * loss_sum[0] / (float)((size_t)N_TOK * C_DIM);
        out3[1] = 0.0f;
        out3[2] = 100.0f * (float)red[0] / (float)V_CODES;
    }
}

// ---------------------------------------------------------------------------
extern "C" void kernel_launch(void* const* d_in, const int* in_sizes, int n_in,
                              void* d_out, int out_size, void* d_ws, size_t ws_size,
                              hipStream_t stream) {
    const float* f   = (const float*)d_in[0];   // [32,64,32,32] f32
    const float* emb = (const float*)d_in[1];   // [8192,64] f32
    float* out = (float*)d_out;                 // fhat (2097152) + 3 scalars

    char* ws = (char*)d_ws;
    _Float16* q      = (_Float16*)(ws + WS_Q_OFF);
    _Float16* wn     = (_Float16*)(ws + WS_WN_OFF);
    int*      idxN   = (int*)     (ws + WS_IDX_OFF);
    unsigned* counts = (unsigned*)(ws + WS_CNT_OFF);
    float*    lsum   = (float*)   (ws + WS_LOSS_OFF);

    vq_zero<<<(V_CODES + 255) / 256, 256, 0, stream>>>(counts, lsum);
    vq_norm_tokens<<<N_TOK / 8, 256, 0, stream>>>(f, q);
    vq_norm_codes<<<V_CODES / 8, 256, 0, stream>>>(emb, wn);
    vq_argmax_wmma<<<N_TOK / 128, 256, 0, stream>>>(q, wn, idxN);
    vq_gather_loss<<<(N_TOK * C_DIM) / 256, 256, 0, stream>>>(f, emb, idxN, out,
                                                              counts, lsum);
    vq_finalize<<<1, 256, 0, stream>>>(counts, lsum, out + (size_t)N_TOK * C_DIM);
}
